// JointModel_12661563589278
// MI455X (gfx1250) — matrix operand order repaired and checked by reading, not compiled, not run on hardware
//
#include <hip/hip_runtime.h>

typedef __bf16 bf16;
typedef __attribute__((ext_vector_type(16))) __bf16 v16bf;
typedef __attribute__((ext_vector_type(8)))  float  v8f;
typedef __attribute__((ext_vector_type(4)))  unsigned v4u;
typedef __attribute__((ext_vector_type(8)))  int      v8i;
typedef __attribute__((ext_vector_type(4)))  int      v4i;

// A-matrix (16x32 bf16) element j -> K index for a lane's half (0: lanes 0-15, 1: lanes 16-31)
__device__ __forceinline__ int wmma_ak(int j, int half) {
  return j + 16 * half;
}

// ---------------------------------------------------------------------------
// TDM: DMA a 64-row x 32-elem bf16 tile (row-major, row pitch = ld elements)
// from global memory into LDS at byte offset lds_off. Remaining extents
// (rows_rem / k_rem) give OOB zero-fill per the D# tensor_dim fields.
// D# layout per CDNA5 ISA ch.8 (group0 128b + group1 256b, 2D tile).
// ---------------------------------------------------------------------------
__device__ __forceinline__ void tdm_load_tile_64x32(const bf16* gbase,
                                                    long row0, long k0, long ld,
                                                    int rows_rem, int k_rem,
                                                    unsigned lds_off)
{
  unsigned long long ga =
      (unsigned long long)(const void*)gbase +
      (unsigned long long)((row0 * ld + k0) * 2);
  v4u g0;
  g0[0] = 1u;                                    // count=1 (valid), no gather
  g0[1] = lds_off;                               // lds_addr (bytes)
  g0[2] = (unsigned)(ga & 0xffffffffull);        // global_addr[31:0]
  g0[3] = (unsigned)((ga >> 32) & 0x1ffffffull)  // global_addr[56:32]
          | (2u << 30);                          // type = 2 ("image")
  v8i g1;
  const int dim0 = k_rem;        // tensor extent along contiguous dim
  const int dim1 = rows_rem;     // tensor extent along rows
  const long s0 = ld;            // tensor_dim0_stride (elements)
  g1[0] = (1 << 16);                                   // data_size = 2 bytes
  g1[1] = (dim0 & 0xffff) << 16;                       // tensor_dim0[15:0]
  g1[2] = ((dim0 >> 16) & 0xffff) | ((dim1 & 0xffff) << 16);
  g1[3] = ((dim1 >> 16) & 0xffff) | (32 << 16);        // tile_dim0 = 32
  g1[4] = 64;                                          // tile_dim1=64, tile_dim2=0
  g1[5] = (int)(s0 & 0xffffffff);                      // stride0[31:0]
  g1[6] = (int)((s0 >> 32) & 0xffff);                  // stride0[47:32], dim1_stride=0
  g1[7] = 0;
  v4i z4 = {};
  v8i z8 = {};
  __builtin_amdgcn_tensor_load_to_lds(g0, g1, z4, z4, z8, 0);
}

// ---------------------------------------------------------------------------
// GEMM: C[M,N] = A[M,K] @ Bt[N,K]^T (both bf16, row-major, K contiguous).
// 128 threads (4 waves), 64x64 output tile, 2x2 waves of 32x32.
// Tiles staged via double-buffered tensor_load_to_lds issued by wave 0,
// gated by s_wait_tensorcnt + workgroup barrier.
// ---------------------------------------------------------------------------
template <typename OT>
__global__ __launch_bounds__(128)
void gemm_tdm_kernel(const bf16* __restrict__ A, int lda, long strideAb,
                     const bf16* __restrict__ Bt, int ldb,
                     OT* __restrict__ C, int ldc, long strideCb,
                     int M, int N, int K)
{
  __shared__ bf16 As[2][64][32];
  __shared__ bf16 Bs[2][64][32];

  const int bz = blockIdx.z;
  A += (long)bz * strideAb;
  C += (long)bz * strideCb;

  const int m0 = blockIdx.y * 64;
  const int n0 = blockIdx.x * 64;
  const int tid = threadIdx.x;
  const int wave = tid >> 5;
  const int lane = tid & 31;
  const int half = lane >> 4;
  const int r    = lane & 15;
  const int wy = wave >> 1, wx = wave & 1;

  const int nk = K >> 5;
  const int rowsA = M - m0;
  const int rowsB = N - n0;

  if (wave == 0) {
    tdm_load_tile_64x32(A, m0, 0, lda, rowsA, K, (unsigned)(size_t)&As[0][0][0]);
    tdm_load_tile_64x32(Bt, n0, 0, ldb, rowsB, K, (unsigned)(size_t)&Bs[0][0][0]);
  }

  v8f acc[2][2] = {};

  for (int kk = 0; kk < nk; kk++) {
    if (wave == 0 && (kk + 1) < nk) {
      const int nb = (kk + 1) & 1;
      const long k0 = (long)(kk + 1) * 32;
      tdm_load_tile_64x32(A, m0, k0, lda, rowsA, K - (int)k0,
                          (unsigned)(size_t)&As[nb][0][0]);
      tdm_load_tile_64x32(Bt, n0, k0, ldb, rowsB, K - (int)k0,
                          (unsigned)(size_t)&Bs[nb][0][0]);
    }
    if ((kk + 1) < nk) __builtin_amdgcn_s_wait_tensorcnt(2);
    else               __builtin_amdgcn_s_wait_tensorcnt(0);
    __syncthreads();

    const int buf = kk & 1;
    v16bf af[2], bf[2];
    #pragma unroll
    for (int i = 0; i < 2; i++) {
      const int rr = wy * 32 + i * 16 + r;
      #pragma unroll
      for (int j = 0; j < 16; j++) af[i][j] = As[buf][rr][wmma_ak(j, half)];
      const int cc = wx * 32 + i * 16 + r;
      #pragma unroll
      for (int j = 0; j < 16; j++) bf[i][j] = Bs[buf][cc][16 * half + j];
    }
    #pragma unroll
    for (int i = 0; i < 2; i++)
      #pragma unroll
      for (int j = 0; j < 2; j++)
        acc[i][j] = __builtin_amdgcn_wmma_f32_16x16x32_bf16(
            false, af[i], false, bf[j], (short)0, acc[i][j], false, false);
    __syncthreads();
  }

  #pragma unroll
  for (int i = 0; i < 2; i++)
    #pragma unroll
    for (int j = 0; j < 2; j++)
      #pragma unroll
      for (int d = 0; d < 8; d++) {
        int gr = m0 + wy * 32 + i * 16 + d + 8 * half;
        int gc = n0 + wx * 32 + j * 16 + r;
        if (gr < M && gc < N) C[(long)gr * ldc + gc] = (OT)acc[i][j][d];
      }
}

// ---------------------------------------------------------------------------
// Pre-passes: f32 -> bf16 copy, and f32 [K][N] -> bf16 [N][K] transpose.
// ---------------------------------------------------------------------------
__global__ __launch_bounds__(256)
void cvt_bf16_kernel(const float* __restrict__ src, bf16* __restrict__ dst, long n)
{
  long i = (long)blockIdx.x * 256 + threadIdx.x;
  if (i < n) dst[i] = (bf16)src[i];
}

__global__ __launch_bounds__(256)
void wtrans_kernel(const float* __restrict__ W, bf16* __restrict__ Wt, int K, int N)
{
  long i = (long)blockIdx.x * 256 + threadIdx.x;
  if (i >= (long)K * N) return;
  int k = (int)(i % K);
  int n = (int)(i / K);
  Wt[i] = (bf16)W[(long)k * N + n];
}

// ---------------------------------------------------------------------------
// RoPE in-place on bf16 Q [b][s][2048] and K [b][s][256].
// grid = (9, 1024, 8): x=0..7 -> Q head, x=8 -> K; block = 128 (pair index d).
// ---------------------------------------------------------------------------
__global__ __launch_bounds__(128)
void rope_kernel(bf16* __restrict__ Q, bf16* __restrict__ Kb,
                 const int* __restrict__ pos0, const int* __restrict__ pos1,
                 const int* __restrict__ pos2)
{
  const int d = threadIdx.x;
  const int s = blockIdx.y;
  const int b = blockIdx.z;
  const int which = blockIdx.x;

  int p;
  if (s < 968)      p = pos0[b * 968 + s];
  else if (s < 974) p = pos1[b * 6 + (s - 968)];
  else              p = pos2[b * 50 + (s - 974)];

  const float inv = __expf(-(float)(2 * d) * (9.210340371976184f / 256.0f));
  const float ang = (float)p * inv;
  const float c = __cosf(ang), sn = __sinf(ang);

  bf16* ptr;
  if (which < 8) ptr = Q + ((long)b * 1024 + s) * 2048 + which * 256;
  else           ptr = Kb + ((long)b * 1024 + s) * 256;

  const float x1 = (float)ptr[d];
  const float x2 = (float)ptr[d + 128];
  ptr[d]       = (bf16)(x1 * c - x2 * sn);
  ptr[d + 128] = (bf16)(x2 * c + x1 * sn);
}

// Vt[b][d][s] = V[b][s][d]   (d=256, s=1024)
__global__ __launch_bounds__(256)
void vtrans_kernel(const bf16* __restrict__ V, bf16* __restrict__ Vt)
{
  long i = (long)blockIdx.x * 256 + threadIdx.x;
  int d = (int)(i & 255);
  long bs = i >> 8;
  int s = (int)(bs & 1023);
  int b = (int)(bs >> 10);
  Vt[((long)b * 256 + d) * 1024 + s] = V[i];
}

// ---------------------------------------------------------------------------
// Flash attention with tanh soft-capping. One wave per (b, h, 16-row q-tile).
// ---------------------------------------------------------------------------
__global__ __launch_bounds__(32)
void attn_kernel(const bf16* __restrict__ Q, const bf16* __restrict__ Kb,
                 const bf16* __restrict__ Vt, const float* __restrict__ mask,
                 bf16* __restrict__ O)
{
  __shared__ bf16 pbuf[16][32];

  const int lane = threadIdx.x;
  const int half = lane >> 4;
  const int r    = lane & 15;
  const int q0 = blockIdx.x * 16;
  const int h  = blockIdx.y;
  const int b  = blockIdx.z;

  const bf16* qbase = Q + ((long)b * 1024 + q0) * 2048 + h * 256;
  v16bf qf[8];
  #pragma unroll
  for (int kc = 0; kc < 8; kc++) {
    const bf16* row = qbase + (long)r * 2048 + kc * 32;
    #pragma unroll
    for (int j = 0; j < 16; j++) qf[kc][j] = row[wmma_ak(j, half)];
  }

  v8f oacc[16] = {};
  float mrow[8], lrow[8];
  #pragma unroll
  for (int d = 0; d < 8; d++) { mrow[d] = -1e30f; lrow[d] = 0.0f; }

  const float scale = 1.0f / (16.0f * 50.0f);  // 1/sqrt(256)/CAP

  for (int n0 = 0; n0 < 1024; n0 += 32) {
    v8f sc[2] = {};
    #pragma unroll
    for (int t = 0; t < 2; t++) {
      #pragma unroll
      for (int kc = 0; kc < 8; kc++) {
        v16bf kf;
        const bf16* krow =
            Kb + ((long)b * 1024 + n0 + t * 16 + r) * 256 + kc * 32 + 16 * half;
        #pragma unroll
        for (int j = 0; j < 16; j++) kf[j] = krow[j];
        sc[t] = __builtin_amdgcn_wmma_f32_16x16x32_bf16(
            false, qf[kc], false, kf, (short)0, sc[t], false, false);
      }
    }

    float p0v[8], p1v[8];
    #pragma unroll
    for (int d = 0; d < 8; d++) {
      const int qr = q0 + d + 8 * half;
      const float* mrowp = mask + ((long)b * 1024 + qr) * 1024 + n0;
      float s0 = tanhf(sc[0][d] * scale) * 50.0f + mrowp[r];
      float s1 = tanhf(sc[1][d] * scale) * 50.0f + mrowp[16 + r];
      float mx = fmaxf(s0, s1);
      #pragma unroll
      for (int off = 1; off < 16; off <<= 1)
        mx = fmaxf(mx, __shfl_xor(mx, off, 32));
      const float mnew = fmaxf(mrow[d], mx);
      const float corr = __expf(mrow[d] - mnew);
      mrow[d] = mnew;
      float p0 = __expf(s0 - mnew);
      float p1 = __expf(s1 - mnew);
      float ls = p0 + p1;
      #pragma unroll
      for (int off = 1; off < 16; off <<= 1) ls += __shfl_xor(ls, off, 32);
      lrow[d] = lrow[d] * corr + ls;
      #pragma unroll
      for (int t = 0; t < 16; t++) oacc[t][d] *= corr;
      p0v[d] = p0; p1v[d] = p1;
    }

    #pragma unroll
    for (int d = 0; d < 8; d++) {
      pbuf[d + 8 * half][r]      = (bf16)p0v[d];
      pbuf[d + 8 * half][r + 16] = (bf16)p1v[d];
    }
    __syncthreads();
    v16bf pf;
    #pragma unroll
    for (int j = 0; j < 16; j++) pf[j] = pbuf[r][wmma_ak(j, half)];
    __syncthreads();

    #pragma unroll
    for (int t = 0; t < 16; t++) {
      v16bf vf;
      const bf16* vrow =
          Vt + ((long)b * 256 + t * 16 + r) * 1024 + n0 + 16 * half;
      #pragma unroll
      for (int j = 0; j < 16; j++) vf[j] = vrow[j];
      oacc[t] = __builtin_amdgcn_wmma_f32_16x16x32_bf16(
          false, pf, false, vf, (short)0, oacc[t], false, false);
    }
  }

  #pragma unroll
  for (int t = 0; t < 16; t++)
    #pragma unroll
    for (int d = 0; d < 8; d++) {
      const int qr = q0 + d + 8 * half;
      const float val = oacc[t][d] / lrow[d];
      O[((long)b * 1024 + qr) * 2048 + h * 256 + t * 16 + r] = (bf16)val;
    }
}

// ---------------------------------------------------------------------------
extern "C" void kernel_launch(void* const* d_in, const int* in_sizes, int n_in,
                              void* d_out, int out_size, void* d_ws, size_t ws_size,
                              hipStream_t stream)
{
  (void)in_sizes; (void)n_in; (void)out_size; (void)ws_size;

  const int Bz = 8, S = 1024;
  const int seg_off[3] = {0, 968, 974};
  const int seg_len[3] = {968, 6, 50};
  const int hidden[3]  = {2048, 1024, 1024};

  const float* hs[3]  = {(const float*)d_in[0], (const float*)d_in[1], (const float*)d_in[2]};
  const float* mask   = (const float*)d_in[3];
  const int*   pos0   = (const int*)d_in[4];
  const int*   pos1   = (const int*)d_in[5];
  const int*   pos2   = (const int*)d_in[6];
  const float* wq[3], *wk[3], *wv[3], *wo[3];
  for (int blk = 0; blk < 3; blk++) {
    wq[blk] = (const float*)d_in[7 + blk * 4 + 0];
    wk[blk] = (const float*)d_in[7 + blk * 4 + 1];
    wv[blk] = (const float*)d_in[7 + blk * 4 + 2];
    wo[blk] = (const float*)d_in[7 + blk * 4 + 3];
  }

  // ---- workspace carve-out (bf16 elements) ----
  bf16* p = (bf16*)d_ws;
  bf16* Qb   = p; p += (long)Bz * S * 2048;     // [8][1024][2048]
  bf16* Kb   = p; p += (long)Bz * S * 256;      // [8][1024][256]
  bf16* Vraw = p; p += (long)Bz * S * 256;
  bf16* Vt   = p; p += (long)Bz * 256 * S;      // [8][256][1024]
  bf16* AO   = p; p += (long)Bz * S * 2048;
  bf16* HSb[3]; bf16* Wqt[3]; bf16* Wkt[3]; bf16* Wvt[3]; bf16* Wot[3];
  for (int blk = 0; blk < 3; blk++) { HSb[blk] = p; p += (long)Bz * seg_len[blk] * hidden[blk]; }
  for (int blk = 0; blk < 3; blk++) { Wqt[blk] = p; p += (long)2048 * hidden[blk]; }   // [2048][hid]
  for (int blk = 0; blk < 3; blk++) { Wkt[blk] = p; p += (long)256 * hidden[blk]; }    // [256][hid]
  for (int blk = 0; blk < 3; blk++) { Wvt[blk] = p; p += (long)256 * hidden[blk]; }
  for (int blk = 0; blk < 3; blk++) { Wot[blk] = p; p += (long)hidden[blk] * 2048; }   // [hid][2048]

  // ---- pre-passes: hs -> bf16; weights -> bf16 transposed [N][K] ----
  for (int blk = 0; blk < 3; blk++) {
    long n = (long)Bz * seg_len[blk] * hidden[blk];
    cvt_bf16_kernel<<<(unsigned)((n + 255) / 256), 256, 0, stream>>>(hs[blk], HSb[blk], n);
    long nq = (long)2048 * hidden[blk];
    wtrans_kernel<<<(unsigned)((nq + 255) / 256), 256, 0, stream>>>(wq[blk], Wqt[blk], hidden[blk], 2048);
    long nkv = (long)256 * hidden[blk];
    wtrans_kernel<<<(unsigned)((nkv + 255) / 256), 256, 0, stream>>>(wk[blk], Wkt[blk], hidden[blk], 256);
    wtrans_kernel<<<(unsigned)((nkv + 255) / 256), 256, 0, stream>>>(wv[blk], Wvt[blk], hidden[blk], 256);
    long no = (long)hidden[blk] * 2048;
    wtrans_kernel<<<(unsigned)((no + 255) / 256), 256, 0, stream>>>(wo[blk], Wot[blk], 2048, hidden[blk]);
  }

  // ---- QKV projections (TDM-staged bf16 WMMA) ----
  for (int blk = 0; blk < 3; blk++) {
    const int M = seg_len[blk], K = hidden[blk];
    dim3 gq((2048 + 63) / 64, (M + 63) / 64, Bz);
    gemm_tdm_kernel<bf16><<<gq, 128, 0, stream>>>(
        HSb[blk], K, (long)M * K, Wqt[blk], K,
        Qb + (long)seg_off[blk] * 2048, 2048, (long)S * 2048, M, 2048, K);
    dim3 gk((256 + 63) / 64, (M + 63) / 64, Bz);
    gemm_tdm_kernel<bf16><<<gk, 128, 0, stream>>>(
        HSb[blk], K, (long)M * K, Wkt[blk], K,
        Kb + (long)seg_off[blk] * 256, 256, (long)S * 256, M, 256, K);
    gemm_tdm_kernel<bf16><<<gk, 128, 0, stream>>>(
        HSb[blk], K, (long)M * K, Wvt[blk], K,
        Vraw + (long)seg_off[blk] * 256, 256, (long)S * 256, M, 256, K);
  }

  // ---- RoPE on Q and K ----
  rope_kernel<<<dim3(9, S, Bz), 128, 0, stream>>>(Qb, Kb, pos0, pos1, pos2);

  // ---- transpose V for attention B-fragments ----
  vtrans_kernel<<<(Bz * S * 256) / 256, 256, 0, stream>>>(Vraw, Vt);

  // ---- attention ----
  attn_kernel<<<dim3(S / 16, 8, Bz), 32, 0, stream>>>(Qb, Kb, Vt, mask, AO);

  // ---- output projections, f32 into concatenated d_out ----
  float* out0 = (float*)d_out;
  float* out1 = out0 + (long)Bz * 968 * 2048;
  float* out2 = out1 + (long)Bz * 6 * 1024;
  float* outs[3] = {out0, out1, out2};
  for (int blk = 0; blk < 3; blk++) {
    const int M = seg_len[blk], N = hidden[blk];
    dim3 go((N + 63) / 64, (M + 63) / 64, Bz);
    gemm_tdm_kernel<float><<<go, 128, 0, stream>>>(
        AO + (long)seg_off[blk] * 2048, 2048, (long)S * 2048, Wot[blk], 2048,
        outs[blk], N, (long)M * N, M, N, 2048);
  }
}
